// EncoderBlock_13245679140888
// MI455X (gfx1250) — matrix operand order repaired and checked by reading, not compiled, not run on hardware
//
#include <hip/hip_runtime.h>
#include <hip/hip_bf16.h>
#include <stdint.h>

typedef __bf16 bf16_t;
typedef __bf16 v16bf __attribute__((ext_vector_type(16)));
typedef float  v8f   __attribute__((ext_vector_type(8)));

#define EMBED   1024
#define DFF     4096
#define NHEADS  16
#define DK      64
#define SEQ     2048
#define BATCH   4
#define ROWS    (BATCH * SEQ)   // 8192

// ---------------------------------------------------------------------------
// LDS byte offset of a shared-memory pointer (AS3 addresses are offsets from
// LDS_BASE, which is what GLOBAL_LOAD_ASYNC_TO_LDS wants in its VDST VGPR).
// ---------------------------------------------------------------------------
__device__ __forceinline__ unsigned lds_offset(void* p) {
    return (unsigned)(uintptr_t)(__attribute__((address_space(3))) void*)p;
}

// Async copy 16 bytes global -> LDS per lane (VGLOBAL GLOBAL_LOAD_ASYNC_TO_LDS_B128,
// tracked with ASYNCcnt). vdst = LDS byte offset, vaddr = 64-bit global address.
__device__ __forceinline__ void async_copy_b128(const void* gp, void* lp) {
    unsigned loff = lds_offset(lp);
    unsigned long long ga = (unsigned long long)(uintptr_t)gp;
    asm volatile("global_load_async_to_lds_b128 %0, %1, off"
                 :: "v"(loff), "v"(ga) : "memory");
}

__device__ __forceinline__ void wait_asynccnt0() {
    asm volatile("s_wait_asynccnt 0x0" ::: "memory");
}

// ---------------------------------------------------------------------------
// Fragment loader: 32 bytes per lane = two ds_load_b128 / global_load_b128.
//   A-operand (16x32 bf16): lane m=lane&15, g=lane>>4; K = g*8+{0..7} then
//   g*8+16+{0..7}  -> gap = 16 elements.
//   B-operand (32x16 bf16, staged K-contiguous per column): lane n=lane&15,
//   g=lane>>4; K = g*8+{0..7} then g*8+16+{0..7} -> gap = 16 elements.
// ---------------------------------------------------------------------------
__device__ __forceinline__ v16bf load_frag(const bf16_t* p, int gap_elems) {
    union { uint4 u[2]; v16bf v; } f;
    f.u[0] = *(const uint4*)(p);
    f.u[1] = *(const uint4*)(p + gap_elems);
    return f.v;
}

__device__ __forceinline__ v8f wmma_bf16(v16bf a, v16bf b, v8f c) {
    return __builtin_amdgcn_wmma_f32_16x16x32_bf16(false, a, false, b,
                                                   (short)0, c, false, false);
}

// ---------------------------------------------------------------------------
// fp32 -> bf16 conversion (grid-stride)
// ---------------------------------------------------------------------------
__global__ __launch_bounds__(256) void cvt_f32_bf16(const float* __restrict__ s,
                                                    bf16_t* __restrict__ d, int n) {
    int i = blockIdx.x * blockDim.x + threadIdx.x;
    int stride = gridDim.x * blockDim.x;
    for (; i < n; i += stride) d[i] = (bf16_t)s[i];
}

// ---------------------------------------------------------------------------
// bf16 GEMM: C[M,N] = op(A[M,K] @ B[K,N] + bias) * scale
// Block tile 128x128, K-step 32, double-buffered LDS.
// A tile staged with global_load_async_to_lds_b128 (row-major, padded 80B
// rows); B tile transposed [n][k] via packed 32-bit LDS stores. 8 waves as
// 4(M) x 2(N); each wave computes 2x4 16x16 WMMA tiles.
// ---------------------------------------------------------------------------
#define LDS_STRIDE 40   // bf16 elements per padded row (80 bytes = 5 * 16B)

template<int RELU>
__global__ __launch_bounds__(256) void gemm_bf16_wmma(
    const bf16_t* __restrict__ A, const bf16_t* __restrict__ B,
    const float* __restrict__ bias, float* __restrict__ outF,
    bf16_t* __restrict__ outH, int M, int N, int K, float scale)
{
    __shared__ alignas(16) bf16_t sA[2][128 * LDS_STRIDE];   // 2 x 10 KB
    __shared__ alignas(16) bf16_t sB[2][128 * LDS_STRIDE];   // 2 x 10 KB (transposed)

    const int tid  = threadIdx.x;
    const int wave = tid >> 5, lane = tid & 31;
    const int wm = wave >> 1, wn = wave & 1;
    const int lm = lane & 15, lg = lane >> 4;
    const int m0 = blockIdx.y * 128, n0 = blockIdx.x * 128;

    v8f acc[2][4];
    #pragma unroll
    for (int i = 0; i < 2; ++i)
        #pragma unroll
        for (int j = 0; j < 4; ++j)
            acc[i][j] = (v8f)(0.0f);

    // --- staging: A via async-to-LDS (2 x 16B per thread), B via register
    //     transpose with packed b32 stores (2 rows x 8 cols per thread).
    auto stage = [&](int buf, int k0) {
        {
            int r = tid >> 2, q = tid & 3;       // rows r, r+64; 8-elem chunk q
            #pragma unroll
            for (int i = 0; i < 2; ++i) {
                int rr = r + i * 64;
                async_copy_b128(A + (size_t)(m0 + rr) * K + k0 + q * 8,
                                &sA[buf][rr * LDS_STRIDE + q * 8]);
            }
        }
        {
            int kp = tid >> 4;                   // k pair 2*kp, 2*kp+1
            int c0 = (tid & 15) * 8;             // 8 columns
            union { uint4 u; unsigned short e[8]; } r0, r1;
            r0.u = *(const uint4*)(B + (size_t)(k0 + 2 * kp) * N + n0 + c0);
            r1.u = *(const uint4*)(B + (size_t)(k0 + 2 * kp + 1) * N + n0 + c0);
            #pragma unroll
            for (int j = 0; j < 8; ++j) {
                unsigned pk = (unsigned)r0.e[j] | ((unsigned)r1.e[j] << 16);
                *(unsigned*)(&sB[buf][(c0 + j) * LDS_STRIDE + 2 * kp]) = pk;
            }
        }
    };

    const int nk = K >> 5;
    stage(0, 0);
    for (int kt = 0; kt < nk; ++kt) {
        wait_asynccnt0();          // our async stage of tile kt is done
        __syncthreads();           // everyone's stage of tile kt visible
        if (kt + 1 < nk) stage((kt + 1) & 1, (kt + 1) * 32);

        const bf16_t* a_base = sA[kt & 1];
        const bf16_t* b_base = sB[kt & 1];
        v16bf af[2], bf[4];
        #pragma unroll
        for (int i = 0; i < 2; ++i)
            af[i] = load_frag(&a_base[(wm * 32 + i * 16 + lm) * LDS_STRIDE + lg * 8], 16);
        #pragma unroll
        for (int j = 0; j < 4; ++j)
            bf[j] = load_frag(&b_base[(wn * 64 + j * 16 + lm) * LDS_STRIDE + lg * 8], 16);

        #pragma unroll
        for (int i = 0; i < 2; ++i)
            #pragma unroll
            for (int j = 0; j < 4; ++j)
                acc[i][j] = wmma_bf16(af[i], bf[j], acc[i][j]);
    }

    // --- epilogue: C layout -> VGPR r holds row (r + 8*(lane>>4)), col lane&15
    #pragma unroll
    for (int i = 0; i < 2; ++i) {
        int row_base = m0 + wm * 32 + i * 16 + lg * 8;
        #pragma unroll
        for (int j = 0; j < 4; ++j) {
            int col = n0 + wn * 64 + j * 16 + lm;
            float bv = bias ? bias[col] : 0.0f;
            #pragma unroll
            for (int r = 0; r < 8; ++r) {
                float v = (acc[i][j][r] + bv) * scale;
                if (RELU) v = fmaxf(v, 0.0f);
                size_t idx = (size_t)(row_base + r) * N + col;
                if (outF) outF[idx] = v;
                if (outH) outH[idx] = (bf16_t)v;
            }
        }
    }
}

// ---------------------------------------------------------------------------
// Flash attention. One WG = (b, h, 128 q rows); wave w owns q rows
// [w*16, w*16+16). Loop over 64-key chunks with online softmax.
// K tile staged via async-to-LDS; V transposed through registers.
// masked positions get 1e-9 (reference quirk); 1/sqrt(dk) folded into Q.
// ---------------------------------------------------------------------------
__global__ __launch_bounds__(256) void attention_wmma(
    const bf16_t* __restrict__ q, const bf16_t* __restrict__ k,
    const bf16_t* __restrict__ v, const int* __restrict__ mask,
    bf16_t* __restrict__ ctx)
{
    __shared__ alignas(16) bf16_t sK[64 * DK];        // [key][dk]   8 KB
    __shared__ alignas(16) bf16_t sV[DK * 64];        // [dk][key]   8 KB (transposed)
    __shared__ alignas(16) bf16_t sP[8][16 * 64];     // per-wave P 16 KB

    const int tid  = threadIdx.x;
    const int wave = tid >> 5, lane = tid & 31;
    const int lm = lane & 15, lg = lane >> 4;
    const int bh = blockIdx.y;
    const int b  = bh / NHEADS, h = bh % NHEADS;
    const int s0 = blockIdx.x * 128;
    const int qrow = s0 + wave * 16;

    // Q fragments straight from global (row-major, dk contiguous): dk=64 -> 2
    v16bf qf[2];
    #pragma unroll
    for (int c = 0; c < 2; ++c)
        qf[c] = load_frag(q + ((size_t)(b * SEQ + qrow + lm)) * EMBED
                            + h * DK + c * 32 + lg * 8, 16);

    v8f o[4];
    #pragma unroll
    for (int j = 0; j < 4; ++j) o[j] = (v8f)(0.0f);
    float mrow[8], lrow[8];
    #pragma unroll
    for (int r = 0; r < 8; ++r) { mrow[r] = -3.0e38f; lrow[r] = 0.0f; }

    for (int kc0 = 0; kc0 < SEQ; kc0 += 64) {
        // stage K tile [64 keys][64 dk] via async copy (row-major is already
        // the B-operand staging layout: dk contiguous per key)
        {
            int key = tid >> 2, part = tid & 3;
            async_copy_b128(k + ((size_t)(b * SEQ + kc0 + key)) * EMBED
                              + h * DK + part * 16,
                            &sK[key * DK + part * 16]);
        }
        // stage V tile transposed [64 dk][64 keys]
        {
            int key = tid >> 2, d0 = (tid & 3) * 16;
            const uint4* g = (const uint4*)(v + ((size_t)(b * SEQ + kc0 + key)) * EMBED
                                              + h * DK + d0);
            union { uint4 u[2]; bf16_t e[16]; } t;
            t.u[0] = g[0]; t.u[1] = g[1];
            #pragma unroll
            for (int j = 0; j < 16; ++j)
                sV[(d0 + j) * 64 + key] = t.e[j];
        }
        wait_asynccnt0();
        __syncthreads();

        // S = Q K^T : 4 key tiles x 2 dk-chunks
        v8f sacc[4];
        #pragma unroll
        for (int nt = 0; nt < 4; ++nt) {
            sacc[nt] = (v8f)(0.0f);
            #pragma unroll
            for (int c = 0; c < 2; ++c) {
                v16bf kf = load_frag(&sK[(nt * 16 + lm) * DK + c * 32 + lg * 8], 16);
                sacc[nt] = wmma_bf16(qf[c], kf, sacc[nt]);
            }
        }

        // mask (reference quirk: masked positions get 1e-9)
        float mv[4];
        #pragma unroll
        for (int nt = 0; nt < 4; ++nt)
            mv[nt] = (float)mask[b * SEQ + kc0 + nt * 16 + lm];

        // online softmax per row (row r lives in VGPR r across 16 lanes of a half)
        #pragma unroll
        for (int r = 0; r < 8; ++r) {
            float mx = -3.0e38f;
            #pragma unroll
            for (int nt = 0; nt < 4; ++nt) {
                float sv = (mv[nt] != 0.0f) ? sacc[nt][r] : 1e-9f;
                sacc[nt][r] = sv;
                mx = fmaxf(mx, sv);
            }
            #pragma unroll
            for (int d = 1; d < 16; d <<= 1) mx = fmaxf(mx, __shfl_xor(mx, d, 32));
            float mnew  = fmaxf(mrow[r], mx);
            float alpha = __expf(mrow[r] - mnew);
            mrow[r] = mnew;
            float rs = 0.0f;
            #pragma unroll
            for (int nt = 0; nt < 4; ++nt) {
                float p = __expf(sacc[nt][r] - mnew);
                sacc[nt][r] = p;
                rs += p;
            }
            #pragma unroll
            for (int d = 1; d < 16; d <<= 1) rs += __shfl_xor(rs, d, 32);
            lrow[r] = lrow[r] * alpha + rs;
            #pragma unroll
            for (int j = 0; j < 4; ++j) o[j][r] *= alpha;
        }

        // P -> per-wave LDS strip (re-fragment into A-operand layout)
        bf16_t* pw = &sP[wave][0];
        #pragma unroll
        for (int nt = 0; nt < 4; ++nt)
            #pragma unroll
            for (int r = 0; r < 8; ++r)
                pw[(r + 8 * lg) * 64 + nt * 16 + lm] = (bf16_t)sacc[nt][r];

        // O += P V : 2 key-chunks x 4 dk tiles
        #pragma unroll
        for (int c = 0; c < 2; ++c) {
            v16bf pf = load_frag(&pw[lm * 64 + c * 32 + lg * 8], 16);
            #pragma unroll
            for (int nt = 0; nt < 4; ++nt) {
                v16bf vf = load_frag(&sV[(nt * 16 + lm) * 64 + c * 32 + lg * 8], 16);
                o[nt] = wmma_bf16(pf, vf, o[nt]);
            }
        }
        __syncthreads();
    }

    // normalize and write ctx (bf16, [B,S,E] with head h at cols h*DK..)
    #pragma unroll
    for (int r = 0; r < 8; ++r) {
        float inv = 1.0f / lrow[r];
        int row = qrow + r + 8 * lg;
        #pragma unroll
        for (int j = 0; j < 4; ++j) {
            int col = h * DK + j * 16 + lm;
            ctx[(size_t)(b * SEQ + row) * EMBED + col] = (bf16_t)(o[j][r] * inv);
        }
    }
}

// ---------------------------------------------------------------------------
// Fused residual + LayerNorm (reference semantics: unbiased std (ddof=1),
// scalar gamma/beta, eps added to std). One WG per row of 1024.
// ---------------------------------------------------------------------------
__global__ __launch_bounds__(256) void add_layernorm(
    const float* __restrict__ x, const float* __restrict__ y,
    const float* __restrict__ gamma, const float* __restrict__ beta,
    float* __restrict__ outF, bf16_t* __restrict__ outH)
{
    __shared__ float redS[8];
    __shared__ float redQ[8];
    const int row = blockIdx.x;
    const float* xr = x + (size_t)row * EMBED;
    const float* yr = y + (size_t)row * EMBED;

    float vals[4], s = 0.0f, ss = 0.0f;
    #pragma unroll
    for (int i = 0; i < 4; ++i) {
        float t = xr[threadIdx.x + i * 256] + yr[threadIdx.x + i * 256];
        vals[i] = t; s += t; ss += t * t;
    }
    #pragma unroll
    for (int d = 1; d < 32; d <<= 1) {
        s  += __shfl_xor(s,  d, 32);
        ss += __shfl_xor(ss, d, 32);
    }
    int wave = threadIdx.x >> 5, lane = threadIdx.x & 31;
    if (lane == 0) { redS[wave] = s; redQ[wave] = ss; }
    __syncthreads();
    float S = 0.0f, SS = 0.0f;
    #pragma unroll
    for (int i = 0; i < 8; ++i) { S += redS[i]; SS += redQ[i]; }

    float mean = S * (1.0f / EMBED);
    float var  = fmaxf((SS - (float)EMBED * mean * mean) * (1.0f / (EMBED - 1)), 0.0f);
    float inv  = gamma[0] / (sqrtf(var) + 1e-6f);
    float bet  = beta[0];
    #pragma unroll
    for (int i = 0; i < 4; ++i) {
        float o = (vals[i] - mean) * inv + bet;
        size_t idx = (size_t)row * EMBED + threadIdx.x + i * 256;
        outF[idx] = o;
        if (outH) outH[idx] = (bf16_t)o;
    }
}

// ---------------------------------------------------------------------------
extern "C" void kernel_launch(void* const* d_in, const int* in_sizes, int n_in,
                              void* d_out, int out_size, void* d_ws, size_t ws_size,
                              hipStream_t stream)
{
    const float* x    = (const float*)d_in[0];
    const int*   mask = (const int*)  d_in[1];
    const float* wq = (const float*)d_in[2];  const float* bq = (const float*)d_in[3];
    const float* wk = (const float*)d_in[4];  const float* bk = (const float*)d_in[5];
    const float* wv = (const float*)d_in[6];  const float* bv = (const float*)d_in[7];
    const float* wo = (const float*)d_in[8];  const float* bo = (const float*)d_in[9];
    const float* w1 = (const float*)d_in[10]; const float* b1 = (const float*)d_in[11];
    const float* w2 = (const float*)d_in[12]; const float* b2 = (const float*)d_in[13];
    const float* gamma = (const float*)d_in[14];
    const float* beta  = (const float*)d_in[15];
    float* out = (float*)d_out;

    char* ws = (char*)d_ws;
    size_t off = 0;
    auto alloc = [&](size_t bytes) { char* p = ws + off; off += bytes; return p; };

    const size_t NE = (size_t)ROWS * EMBED;           // 8192*1024
    bf16_t* xb   = (bf16_t*)alloc(NE * 2);
    bf16_t* wqb  = (bf16_t*)alloc((size_t)EMBED * EMBED * 2);
    bf16_t* wkb  = (bf16_t*)alloc((size_t)EMBED * EMBED * 2);
    bf16_t* wvb  = (bf16_t*)alloc((size_t)EMBED * EMBED * 2);
    bf16_t* wob  = (bf16_t*)alloc((size_t)EMBED * EMBED * 2);
    bf16_t* w1b  = (bf16_t*)alloc((size_t)EMBED * DFF * 2);
    bf16_t* w2b  = (bf16_t*)alloc((size_t)DFF * EMBED * 2);
    bf16_t* qb   = (bf16_t*)alloc(NE * 2);
    bf16_t* kb   = (bf16_t*)alloc(NE * 2);
    bf16_t* vb   = (bf16_t*)alloc(NE * 2);
    bf16_t* ctxb = (bf16_t*)alloc(NE * 2);
    bf16_t* ff1b = qb;  // alias: q/k/v/ctx region (64 MB) is dead by FFN time
    bf16_t* h1b  = (bf16_t*)alloc(NE * 2);
    float*  attnF = (float*)alloc(NE * 4);            // reused as ff2 output
    float*  h1f   = (float*)alloc(NE * 4);

    dim3 blk(256);
    dim3 cgrid(1024);
    cvt_f32_bf16<<<cgrid, blk, 0, stream>>>(x,  xb,  (int)NE);
    cvt_f32_bf16<<<cgrid, blk, 0, stream>>>(wq, wqb, EMBED * EMBED);
    cvt_f32_bf16<<<cgrid, blk, 0, stream>>>(wk, wkb, EMBED * EMBED);
    cvt_f32_bf16<<<cgrid, blk, 0, stream>>>(wv, wvb, EMBED * EMBED);
    cvt_f32_bf16<<<cgrid, blk, 0, stream>>>(wo, wob, EMBED * EMBED);
    cvt_f32_bf16<<<cgrid, blk, 0, stream>>>(w1, w1b, EMBED * DFF);
    cvt_f32_bf16<<<cgrid, blk, 0, stream>>>(w2, w2b, DFF * EMBED);

    // QKV projections (1/sqrt(dk)=0.125 folded into Q)
    dim3 ggrid(EMBED / 128, ROWS / 128);   // (8, 64)
    gemm_bf16_wmma<0><<<ggrid, blk, 0, stream>>>(xb, wqb, bq, nullptr, qb,
                                                 ROWS, EMBED, EMBED, 0.125f);
    gemm_bf16_wmma<0><<<ggrid, blk, 0, stream>>>(xb, wkb, bk, nullptr, kb,
                                                 ROWS, EMBED, EMBED, 1.0f);
    gemm_bf16_wmma<0><<<ggrid, blk, 0, stream>>>(xb, wvb, bv, nullptr, vb,
                                                 ROWS, EMBED, EMBED, 1.0f);

    // Attention
    dim3 agrid(SEQ / 128, BATCH * NHEADS); // (16, 64)
    attention_wmma<<<agrid, blk, 0, stream>>>(qb, kb, vb, mask, ctxb);

    // Output projection
    gemm_bf16_wmma<0><<<ggrid, blk, 0, stream>>>(ctxb, wob, bo, attnF, nullptr,
                                                 ROWS, EMBED, EMBED, 1.0f);

    // Residual + LN 1
    add_layernorm<<<dim3(ROWS), blk, 0, stream>>>(x, attnF, gamma, beta, h1f, h1b);

    // FFN
    dim3 fgrid1(DFF / 128, ROWS / 128);    // (32, 64)
    gemm_bf16_wmma<1><<<fgrid1, blk, 0, stream>>>(h1b, w1b, b1, nullptr, ff1b,
                                                  ROWS, DFF, EMBED, 1.0f);
    gemm_bf16_wmma<0><<<ggrid, blk, 0, stream>>>(ff1b, w2b, b2, attnF, nullptr,
                                                 ROWS, EMBED, DFF, 1.0f);

    // Residual + LN 2 -> final output
    add_layernorm<<<dim3(ROWS), blk, 0, stream>>>(h1f, attnF, gamma, beta, out, nullptr);
}
